// StructureEncoder_78675210928263
// MI455X (gfx1250) — matrix operand order repaired and checked by reading, not compiled, not run on hardware
//
#include <hip/hip_runtime.h>
#include <math.h>

// ---------------- problem constants ----------------
#define B_      8
#define L_      512
#define KNN_    16
#define D_      1024
#define H_      16
#define NL_     2
#define DLAT_   64
#define DH_     64
#define DFF_    4096
#define HENC_   512
#define NTOK_   (B_*L_*KNN_)   // 65536 tokens through the transformer
#define NNODE_  (B_*L_)        // 4096 nodes

typedef __attribute__((ext_vector_type(16))) __bf16 v16bf;
typedef __attribute__((ext_vector_type(8)))  __bf16 v8bf;
typedef __attribute__((ext_vector_type(8)))  float  v8f;
typedef unsigned int u32x4 __attribute__((ext_vector_type(4)));
typedef int          i32x4 __attribute__((ext_vector_type(4)));
typedef int          i32x8 __attribute__((ext_vector_type(8)));

typedef __attribute__((address_space(1))) void gvoid;    // global
typedef __attribute__((address_space(3))) void lvoid;    // LDS
typedef __attribute__((address_space(1))) i32x4 as1_i32x4;  // global int4
typedef __attribute__((address_space(3))) i32x4 as3_i32x4;  // LDS int4

// ---- CDNA5 async / TDM capability probes (compile-time) ----
#if __has_builtin(__builtin_amdgcn_global_load_async_to_lds_b128) && \
    __has_builtin(__builtin_amdgcn_s_wait_asynccnt)
#define HAVE_ASYNC 1
#else
#define HAVE_ASYNC 0
#endif
#if HAVE_ASYNC && __has_builtin(__builtin_amdgcn_tensor_load_to_lds) && \
    __has_builtin(__builtin_amdgcn_s_wait_tensorcnt)
#define HAVE_TDM 1
#else
#define HAVE_TDM 0
#endif

union FragBF { v16bf v; v8bf h[2]; };

__device__ __forceinline__ float bf2f(unsigned short u) {
  union { unsigned int ui; float fl; } c; c.ui = ((unsigned int)u) << 16; return c.fl;
}
__device__ __forceinline__ unsigned short f2bf(float f) {
  union { unsigned int ui; float fl; } c; c.fl = f;
  unsigned int r = c.ui + 0x7FFFu + ((c.ui >> 16) & 1u);
  return (unsigned short)(r >> 16);
}

// ---------------- weight convert + transpose: W[K][N] f32 -> Wt[N][K] bf16 ----
__global__ void __launch_bounds__(256)
convert_wT_kernel(const float* __restrict__ W, unsigned short* __restrict__ Wt,
                  int K, int N) {
  long idx = (long)blockIdx.x * 256 + threadIdx.x;
  if (idx >= (long)K * N) return;
  int n = (int)(idx / K);
  int k = (int)(idx - (long)n * K);
  Wt[idx] = f2bf(W[(size_t)k * N + n]);
}

// ---------------- KNN: per (b,i) pick 16 smallest masked d2 (ties -> low idx) --
__global__ void __launch_bounds__(256)
knn_kernel(const float* __restrict__ coords, const unsigned char* __restrict__ am,
           const int* __restrict__ sid, int* __restrict__ edges) {
  int n = blockIdx.x;
  int b = n >> 9;
  int i = n & 511;
  int tid = threadIdx.x;
  __shared__ float sval[512];
  __shared__ float rv[256];
  __shared__ int   ri[256];
  size_t ibase = ((size_t)(b * L_ + i) * 3 + 1) * 3;
  float cx = coords[ibase + 0], cy = coords[ibase + 1], cz = coords[ibase + 2];
  int si = sid[b * L_ + i];
  for (int j = tid; j < 512; j += 256) {
    size_t jb = ((size_t)(b * L_ + j) * 3 + 1) * 3;
    float dx = coords[jb + 0] - cx, dy = coords[jb + 1] - cy, dz = coords[jb + 2] - cz;
    float d2 = dx * dx + dy * dy + dz * dz;
    bool valid = (am[b * L_ + j] != 0) && (sid[b * L_ + j] == si);
    sval[j] = valid ? d2 : 3.0e38f;
  }
  __syncthreads();
  for (int it = 0; it < KNN_; ++it) {
    float v0 = sval[tid];       int i0 = tid;
    float v1 = sval[tid + 256]; int i1 = tid + 256;
    if (v1 < v0) { v0 = v1; i0 = i1; }
    rv[tid] = v0; ri[tid] = i0;
    __syncthreads();
    for (int s = 128; s > 0; s >>= 1) {
      if (tid < s) {
        float ov = rv[tid + s]; int oi = ri[tid + s];
        if (ov < rv[tid] || (ov == rv[tid] && oi < ri[tid])) { rv[tid] = ov; ri[tid] = oi; }
      }
      __syncthreads();
    }
    if (tid == 0) { edges[n * KNN_ + it] = ri[0]; sval[ri[0]] = 3.0e38f; }
    __syncthreads();
  }
}

// ---------------- init: z = emb[clip(diff)], dist matrix --------------------
__global__ void __launch_bounds__(256)
init_kernel(const float* __restrict__ coords, const int* __restrict__ edges,
            const float* __restrict__ emb, float* __restrict__ z,
            float* __restrict__ dist) {
  int n = blockIdx.x;
  int b = n >> 9;
  int tid = threadIdx.x;
  __shared__ int sE[KNN_];
  __shared__ float px[KNN_], py[KNN_], pz[KNN_];
  if (tid < KNN_) {
    int e = edges[n * KNN_ + tid];
    sE[tid] = e;
    size_t base = ((size_t)(b * L_ + e) * 3 + 1) * 3;
    px[tid] = coords[base + 0]; py[tid] = coords[base + 1]; pz[tid] = coords[base + 2];
  }
  __syncthreads();
  {
    int e = tid >> 4, f = tid & 15;
    float dx = px[e] - px[f], dy = py[e] - py[f], dz = pz[e] - pz[f];
    dist[(size_t)n * 256 + tid] = sqrtf(dx * dx + dy * dy + dz * dz + 1e-8f);
  }
  int e0 = sE[0];
  for (int idx = tid; idx < KNN_ * D_; idx += 256) {
    int e = idx >> 10, d = idx & 1023;
    int diff = sE[e] - e0;
    diff = diff < -32 ? -32 : (diff > 32 ? 32 : diff);
    z[((size_t)n * KNN_ + e) * D_ + d] = emb[(size_t)(diff + 32) * D_ + d];
  }
}

// ---------------- LayerNorm f32 -> bf16 (optional row mask-zero) -------------
__global__ void __launch_bounds__(256)
ln_kernel(const float* __restrict__ X, long rowStride,
          const float* __restrict__ g, const float* __restrict__ bt,
          unsigned short* __restrict__ out, const unsigned char* __restrict__ mask) {
  int row = blockIdx.x;
  int tid = threadIdx.x;
  const float* x = X + (size_t)row * rowStride;
  __shared__ float s1[256], s2[256];
  float a = 0.f, sq = 0.f, v[4];
  for (int i = 0; i < 4; ++i) {
    float t = x[tid + 256 * i];
    v[i] = t; a += t; sq += t * t;
  }
  s1[tid] = a; s2[tid] = sq;
  __syncthreads();
  for (int s = 128; s > 0; s >>= 1) {
    if (tid < s) { s1[tid] += s1[tid + s]; s2[tid] += s2[tid + s]; }
    __syncthreads();
  }
  float mean = s1[0] * (1.0f / D_);
  float var = s2[0] * (1.0f / D_) - mean * mean;
  float rinv = rsqrtf(var + 1e-5f);
  bool keep = (mask == nullptr) || (mask[row] != 0);
  unsigned short* o = out + (size_t)row * D_;
  for (int i = 0; i < 4; ++i) {
    int c = tid + 256 * i;
    float y = keep ? ((v[i] - mean) * rinv * g[c] + bt[c]) : 0.0f;
    o[c] = f2bf(y);
  }
}

// ---------------- tiled bf16 WMMA GEMM, double-buffered async staging --------
// out = A[MxK] * Bt[NxK]^T + bias
// MODE 0: outBf = bf16(acc + bias)
// MODE 1: outF  = acc + bias + resid   (fp32, residual stream)
// MODE 2: outF  = acc + bias           (fp32)
template <int MODE>
__global__ void __launch_bounds__(256)
gemm_bf16_kernel(const unsigned short* __restrict__ A,
                 const unsigned short* __restrict__ Bt,
                 const float* __restrict__ bias,
                 const float* __restrict__ resid,
                 float* __restrict__ outF,
                 unsigned short* __restrict__ outBf,
                 int M, int N, int K) {
  __shared__ __align__(16) unsigned short sA[2][128 * 64];
  __shared__ __align__(16) unsigned short sB[2][128 * 64];
  const int tid  = threadIdx.x;
  const int lane = tid & 31;
  const int wid  = tid >> 5;          // 8 waves
  const int wm   = (wid >> 1) * 32;   // 4 waves down M
  const int wn   = (wid & 1) * 64;    // 2 waves across N
  const int m0   = blockIdx.y * 128;
  const int n0   = blockIdx.x * 128;

  v8f acc[2][4] = {};

#if HAVE_ASYNC
  // Issue one stage of async staging: A tile always via per-lane async-to-LDS;
  // B (weight) tile via the Tensor Data Mover when available.
  auto issueStage = [&](int buf, int k0) {
    for (int idx = tid; idx < 1024; idx += 256) {
      int r = idx >> 3, seg = idx & 7;
      __builtin_amdgcn_global_load_async_to_lds_b128(
          (as1_i32x4*)&A[(size_t)(m0 + r) * K + k0 + seg * 8],
          (as3_i32x4*)&sA[buf][r * 64 + seg * 8], 0, 0);
#if !HAVE_TDM
      __builtin_amdgcn_global_load_async_to_lds_b128(
          (as1_i32x4*)&Bt[(size_t)(n0 + r) * K + k0 + seg * 8],
          (as3_i32x4*)&sB[buf][r * 64 + seg * 8], 0, 0);
#endif
    }
#if HAVE_TDM
    if (wid == 0) {
      // D# group0: count=1, lds_addr, 57b global addr, type=2 ("image")
      unsigned ldsAddr = (unsigned)(unsigned long long)(lvoid*)&sB[buf][0];
      unsigned long long ga =
          (unsigned long long)(const void*)&Bt[(size_t)n0 * K + k0];
      u32x4 g0;
      g0.x = 1u;
      g0.y = ldsAddr;
      g0.z = (unsigned)(ga & 0xFFFFFFFFull);
      g0.w = (unsigned)((ga >> 32) & 0x01FFFFFFull) | (2u << 30);
      // D# group1: data_size=2B, tensor_dim0=K, tensor_dim1=N,
      //            tile 64 x 128, tensor_dim0_stride=K
      i32x8 g1;
      g1[0] = (1 << 16);                                    // data_size = 2B
      g1[1] = (int)(((unsigned)K & 0xFFFFu) << 16);         // tensor_dim0 lo
      g1[2] = (int)((((unsigned)K >> 16) & 0xFFFFu) |
                    (((unsigned)N & 0xFFFFu) << 16));       // dim0 hi | dim1 lo
      g1[3] = (int)((((unsigned)N >> 16) & 0xFFFFu) |
                    (64u << 16));                           // dim1 hi | tile_dim0
      g1[4] = 128;                                          // tile_dim1
      g1[5] = K;                                            // dim0_stride lo32
      g1[6] = 0;
      g1[7] = 0;
      i32x4 gz = {0, 0, 0, 0};
#if __clang_major__ >= 23
      i32x8 gz8 = {0, 0, 0, 0, 0, 0, 0, 0};
      __builtin_amdgcn_tensor_load_to_lds(g0, g1, gz, gz, gz8, 0);
#else
      __builtin_amdgcn_tensor_load_to_lds(g0, g1, gz, gz, 0);
#endif
    }
#endif
  };
#endif  // HAVE_ASYNC

  // WMMA compute on one staged buffer
  auto computeStage = [&](int buf) {
    const unsigned short* cA = &sA[buf][0];
    const unsigned short* cB = &sB[buf][0];
    #pragma unroll
    for (int kk = 0; kk < 64; kk += 32) {
      FragBF afrag[2], bfrag[4];
      const int klo = (lane < 16) ? 0 : 8;       // A: K 0-7/16-23 vs 8-15/24-31
      #pragma unroll
      for (int mi = 0; mi < 2; ++mi) {
        int mrow = wm + mi * 16 + (lane & 15);
        afrag[mi].h[0] = *(const v8bf*)&cA[mrow * 64 + kk + klo];
        afrag[mi].h[1] = *(const v8bf*)&cA[mrow * 64 + kk + 16 + klo];
      }
      const int kboff = kk + ((lane < 16) ? 0 : 8);
      #pragma unroll
      for (int ni = 0; ni < 4; ++ni) {
        int ncol = wn + ni * 16 + (lane & 15);
        bfrag[ni].h[0] = *(const v8bf*)&cB[ncol * 64 + kboff];
        bfrag[ni].h[1] = *(const v8bf*)&cB[ncol * 64 + kboff + 16];
      }
      #pragma unroll
      for (int mi = 0; mi < 2; ++mi)
        #pragma unroll
        for (int ni = 0; ni < 4; ++ni)
          acc[mi][ni] = __builtin_amdgcn_wmma_f32_16x16x32_bf16(
              false, afrag[mi].v, false, bfrag[ni].v,
              (short)0, acc[mi][ni], false, false);
    }
  };

#if HAVE_ASYNC
  issueStage(0, 0);
  int s = 0;
  for (int k0 = 0; k0 < K; k0 += 64, s ^= 1) {
    const bool hasNext = (k0 + 64 < K);
    if (hasNext) issueStage(s ^ 1, k0 + 64);
    // wait for the CURRENT stage's async loads (in-order per wave)
#if HAVE_TDM
    if (hasNext) __builtin_amdgcn_s_wait_asynccnt(4);
    else         __builtin_amdgcn_s_wait_asynccnt(0);
    if (wid == 0) {
      if (hasNext) __builtin_amdgcn_s_wait_tensorcnt(1);
      else         __builtin_amdgcn_s_wait_tensorcnt(0);
    }
#else
    if (hasNext) __builtin_amdgcn_s_wait_asynccnt(8);
    else         __builtin_amdgcn_s_wait_asynccnt(0);
#endif
    __syncthreads();
    computeStage(s);
    __syncthreads();    // stage s is free for reuse before next issue
  }
#else
  for (int k0 = 0; k0 < K; k0 += 64) {
    for (int idx = tid; idx < 1024; idx += 256) {
      int row = idx >> 3, seg = idx & 7;
      *(uint4*)&sA[0][row * 64 + seg * 8] =
          *(const uint4*)&A[(size_t)(m0 + row) * K + k0 + seg * 8];
      *(uint4*)&sB[0][row * 64 + seg * 8] =
          *(const uint4*)&Bt[(size_t)(n0 + row) * K + k0 + seg * 8];
    }
    if (k0 + 64 < K) {
      __builtin_prefetch(&A[(size_t)(m0 + (tid >> 1)) * K + k0 + 64], 0, 1);
      __builtin_prefetch(&Bt[(size_t)(n0 + (tid >> 1)) * K + k0 + 64], 0, 1);
    }
    __syncthreads();
    computeStage(0);
    __syncthreads();
  }
#endif

  // epilogue per §7.12.2: VGPR r -> M = r (+8 for upper lane half), N = lane&15
  const int rbump = (lane < 16) ? 0 : 8;
  const int col_l = lane & 15;
  #pragma unroll
  for (int mi = 0; mi < 2; ++mi) {
    #pragma unroll
    for (int ni = 0; ni < 4; ++ni) {
      int colg = n0 + wn + ni * 16 + col_l;
      float bv = bias[colg];
      #pragma unroll
      for (int r = 0; r < 8; ++r) {
        int rowg = m0 + wm + mi * 16 + r + rbump;
        size_t o = (size_t)rowg * N + colg;
        float v = acc[mi][ni][r] + bv;
        if (MODE == 1) outF[o] = v + resid[o];
        else if (MODE == 2) outF[o] = v;
        else outBf[o] = f2bf(v);
      }
    }
  }
}

// ---------------- local KNN attention (16x16 per head, dist bias) ------------
__global__ void __launch_bounds__(256)
attn_kernel(const unsigned short* __restrict__ qkv, const int* __restrict__ edges,
            const unsigned char* __restrict__ am, const float* __restrict__ dist,
            const float* __restrict__ dist_scale_l, unsigned short* __restrict__ obf) {
  int n = blockIdx.x;
  int b = n >> 9;
  int tid = threadIdx.x;
  __shared__ __align__(16) unsigned short kLds[KNN_ * 1024]; // K rows, all heads
  __shared__ unsigned char sM[KNN_];
  if (tid < KNN_) sM[tid] = am[b * L_ + edges[n * KNN_ + tid]];
#if HAVE_ASYNC
  for (int idx = tid; idx < KNN_ * 1024 / 8; idx += 256) {
    int tok = idx >> 7, seg = idx & 127;
    __builtin_amdgcn_global_load_async_to_lds_b128(
        (as1_i32x4*)&qkv[(size_t)(n * KNN_ + tok) * 3072 + 1024 + seg * 8],
        (as3_i32x4*)&kLds[tok * 1024 + seg * 8], 0, 0);
  }
  __builtin_amdgcn_s_wait_asynccnt(0);
#else
  for (int idx = tid; idx < KNN_ * 1024 / 8; idx += 256) {
    int tok = idx >> 7, seg = idx & 127;
    *(uint4*)&kLds[tok * 1024 + seg * 8] =
        *(const uint4*)&qkv[(size_t)(n * KNN_ + tok) * 3072 + 1024 + seg * 8];
  }
#endif
  __syncthreads();
  int wave = tid >> 5, lane = tid & 31;
  int h = wave * 2 + (lane >> 4);
  int e = lane & 15;
  float q[DH_];
  const unsigned short* qp = qkv + (size_t)(n * KNN_ + e) * 3072 + h * DH_;
  #pragma unroll 8
  for (int d = 0; d < DH_; ++d) q[d] = bf2f(qp[d]);
  float xs = dist_scale_l[h];
  float sp = (xs > 20.f) ? xs : log1pf(expf(xs));
  float s[KNN_];
  float mx = -1e30f;
  for (int f = 0; f < KNN_; ++f) {
    float a = 0.f;
    const unsigned short* kp = &kLds[f * 1024 + h * DH_];
    #pragma unroll 8
    for (int d = 0; d < DH_; ++d) a += q[d] * bf2f(kp[d]);
    a = a * 0.125f - sp * dist[(size_t)n * 256 + e * 16 + f];
    if (!sM[f]) a = -1.0e9f;
    s[f] = a;
    mx = fmaxf(mx, a);
  }
  float sum = 0.f;
  for (int f = 0; f < KNN_; ++f) { s[f] = expf(s[f] - mx); sum += s[f]; }
  float inv = 1.f / sum;
  unsigned short* op = obf + (size_t)(n * KNN_ + e) * D_ + h * DH_;
  for (int d = 0; d < DH_; ++d) {
    float a = 0.f;
    for (int f = 0; f < KNN_; ++f)
      a += s[f] * bf2f(qkv[(size_t)(n * KNN_ + f) * 3072 + 2048 + h * DH_ + d]);
    op[d] = f2bf(a * inv);
  }
}

// ---------------- gated SiLU: out = silu(in[:,:h]) * in[:,h:2h] --------------
__global__ void __launch_bounds__(256)
gate_kernel(const unsigned short* __restrict__ in, unsigned short* __restrict__ out,
            int halfN, long rows) {
  size_t idx = (size_t)blockIdx.x * 256 + threadIdx.x;
  if (idx >= (size_t)rows * halfN) return;
  size_t m = idx / halfN;
  int j = (int)(idx - m * halfN);
  float a = bf2f(in[m * (2 * (size_t)halfN) + j]);
  float g = bf2f(in[m * (2 * (size_t)halfN) + halfN + j]);
  float sil = a / (1.f + expf(-a));
  out[idx] = f2bf(sil * g);
}

// ---------------- reparameterize + pack outputs ------------------------------
__global__ void __launch_bounds__(256)
reparam_kernel(const float* __restrict__ e2, const float* __restrict__ eps,
               float* __restrict__ out) {
  int idx = blockIdx.x * 256 + threadIdx.x;
  if (idx >= NNODE_ * DLAT_) return;
  int n = idx >> 6, d = idx & 63;
  float mu = e2[n * 128 + d];
  float lv = fminf(e2[n * 128 + 64 + d], 5.0f);
  float zs = mu + eps[idx] * expf(0.5f * lv);
  out[idx] = zs;
  out[NNODE_ * DLAT_ + idx] = mu;
  out[2 * NNODE_ * DLAT_ + idx] = lv;
}

// ============================ host orchestration =============================
extern "C" void kernel_launch(void* const* d_in, const int* in_sizes, int n_in,
                              void* d_out, int out_size, void* d_ws, size_t ws_size,
                              hipStream_t stream) {
  (void)in_sizes; (void)n_in; (void)out_size; (void)ws_size;
  const float* coords  = (const float*)d_in[0];
  const float* eps     = (const float*)d_in[1];
  const float* emb     = (const float*)d_in[2];
  const float* ln1_g   = (const float*)d_in[3];
  const float* ln1_b   = (const float*)d_in[4];
  const float* Wqkv    = (const float*)d_in[5];
  const float* bqkv    = (const float*)d_in[6];
  const float* dscale  = (const float*)d_in[7];
  const float* Wo      = (const float*)d_in[8];
  const float* bo      = (const float*)d_in[9];
  const float* ln2_g   = (const float*)d_in[10];
  const float* ln2_b   = (const float*)d_in[11];
  const float* Wf1     = (const float*)d_in[12];
  const float* bf1     = (const float*)d_in[13];
  const float* Wf2     = (const float*)d_in[14];
  const float* bf2     = (const float*)d_in[15];
  const float* lnf_g   = (const float*)d_in[16];
  const float* lnf_b   = (const float*)d_in[17];
  const float* We1     = (const float*)d_in[18];
  const float* be1     = (const float*)d_in[19];
  const float* We2     = (const float*)d_in[20];
  const float* be2     = (const float*)d_in[21];
  const unsigned char* am = (const unsigned char*)d_in[22];
  const int* sid       = (const int*)d_in[23];

  char* ws = (char*)d_ws;
  size_t off = 0;
  auto alloc = [&](size_t bytes) -> void* {
    void* p = ws + off;
    off += (bytes + 255) & ~(size_t)255;
    return p;
  };
  float*          z      = (float*)alloc((size_t)NTOK_ * D_ * 4);
  unsigned short* abf    = (unsigned short*)alloc((size_t)NTOK_ * D_ * 2);
  unsigned short* obf    = (unsigned short*)alloc((size_t)NTOK_ * D_ * 2);
  unsigned short* qkvbf  = (unsigned short*)alloc((size_t)NTOK_ * 3072 * 2);
  unsigned short* f1bf   = (unsigned short*)alloc((size_t)NTOK_ * 8192 * 2);
  unsigned short* gbf    = (unsigned short*)alloc((size_t)NTOK_ * 4096 * 2);
  unsigned short* WqkvT  = (unsigned short*)alloc((size_t)NL_ * D_ * 3072 * 2);
  unsigned short* WoT    = (unsigned short*)alloc((size_t)NL_ * D_ * D_ * 2);
  unsigned short* Wf1T   = (unsigned short*)alloc((size_t)NL_ * D_ * 8192 * 2);
  unsigned short* Wf2T   = (unsigned short*)alloc((size_t)NL_ * 4096 * D_ * 2);
  unsigned short* We1T   = (unsigned short*)alloc((size_t)D_ * 1024 * 2);
  unsigned short* We2T   = (unsigned short*)alloc((size_t)HENC_ * 128 * 2);
  int*            edges  = (int*)alloc((size_t)NNODE_ * KNN_ * 4);
  float*          dist   = (float*)alloc((size_t)NNODE_ * 256 * 4);
  unsigned short* zcbf   = (unsigned short*)alloc((size_t)NNODE_ * D_ * 2);
  unsigned short* e1bf   = (unsigned short*)alloc((size_t)NNODE_ * 1024 * 2);
  unsigned short* hhbf   = (unsigned short*)alloc((size_t)NNODE_ * HENC_ * 2);
  float*          e2f    = (float*)alloc((size_t)NNODE_ * 128 * 4);

  auto cvt = [&](const float* W, unsigned short* Wt, int K, int N) {
    long tot = (long)K * N;
    convert_wT_kernel<<<dim3((unsigned)((tot + 255) / 256)), 256, 0, stream>>>(W, Wt, K, N);
  };
  for (int l = 0; l < NL_; ++l) {
    cvt(Wqkv + (size_t)l * D_ * 3072, WqkvT + (size_t)l * 3072 * D_, D_, 3072);
    cvt(Wo   + (size_t)l * D_ * D_,   WoT   + (size_t)l * D_ * D_,   D_, D_);
    cvt(Wf1  + (size_t)l * D_ * 8192, Wf1T  + (size_t)l * 8192 * D_, D_, 8192);
    cvt(Wf2  + (size_t)l * 4096 * D_, Wf2T  + (size_t)l * D_ * 4096, 4096, D_);
  }
  cvt(We1, We1T, D_, 1024);
  cvt(We2, We2T, HENC_, 128);

  knn_kernel<<<NNODE_, 256, 0, stream>>>(coords, am, sid, edges);
  init_kernel<<<NNODE_, 256, 0, stream>>>(coords, edges, emb, z, dist);

  for (int l = 0; l < NL_; ++l) {
    const unsigned short* wq = WqkvT + (size_t)l * 3072 * D_;
    const unsigned short* wo = WoT   + (size_t)l * D_ * D_;
    const unsigned short* w1 = Wf1T  + (size_t)l * 8192 * D_;
    const unsigned short* w2 = Wf2T  + (size_t)l * D_ * 4096;

    ln_kernel<<<NTOK_, 256, 0, stream>>>(z, D_, ln1_g + l * D_, ln1_b + l * D_, abf, nullptr);
    gemm_bf16_kernel<0><<<dim3(3072 / 128, NTOK_ / 128), 256, 0, stream>>>(
        abf, wq, bqkv + (size_t)l * 3072, nullptr, nullptr, qkvbf, NTOK_, 3072, D_);
    attn_kernel<<<NNODE_, 256, 0, stream>>>(qkvbf, edges, am, dist, dscale + l * H_, obf);
    gemm_bf16_kernel<1><<<dim3(D_ / 128, NTOK_ / 128), 256, 0, stream>>>(
        obf, wo, bo + (size_t)l * D_, z, z, nullptr, NTOK_, D_, D_);
    ln_kernel<<<NTOK_, 256, 0, stream>>>(z, D_, ln2_g + l * D_, ln2_b + l * D_, abf, nullptr);
    gemm_bf16_kernel<0><<<dim3(8192 / 128, NTOK_ / 128), 256, 0, stream>>>(
        abf, w1, bf1 + (size_t)l * 8192, nullptr, nullptr, f1bf, NTOK_, 8192, D_);
    {
      size_t tot = (size_t)NTOK_ * 4096;
      gate_kernel<<<dim3((unsigned)((tot + 255) / 256)), 256, 0, stream>>>(f1bf, gbf, 4096, NTOK_);
    }
    gemm_bf16_kernel<1><<<dim3(D_ / 128, NTOK_ / 128), 256, 0, stream>>>(
        gbf, w2, bf2 + (size_t)l * D_, z, z, nullptr, NTOK_, D_, 4096);
  }

  ln_kernel<<<NNODE_, 256, 0, stream>>>(z, (long)KNN_ * D_, lnf_g, lnf_b, zcbf, am);
  gemm_bf16_kernel<0><<<dim3(1024 / 128, NNODE_ / 128), 256, 0, stream>>>(
      zcbf, We1T, be1, nullptr, nullptr, e1bf, NNODE_, 1024, D_);
  {
    size_t tot = (size_t)NNODE_ * HENC_;
    gate_kernel<<<dim3((unsigned)((tot + 255) / 256)), 256, 0, stream>>>(e1bf, hhbf, HENC_, NNODE_);
  }
  gemm_bf16_kernel<2><<<dim3(128 / 128, NNODE_ / 128), 256, 0, stream>>>(
      hhbf, We2T, be2, nullptr, e2f, nullptr, NNODE_, 128, HENC_);
  reparam_kernel<<<dim3((NNODE_ * DLAT_ + 255) / 256), 256, 0, stream>>>(
      e2f, eps, (float*)d_out);
}
